// SAE_topk_28389733827292
// MI455X (gfx1250) — matrix operand order repaired and checked by reading, not compiled, not run on hardware
//
#include <hip/hip_runtime.h>
#include <stdint.h>

#define TOKENS 16384
#define IN     768
#define HID    24576
#define TOPK   32
#define TT     64          // tokens per workgroup (4 WMMA B-tiles per A fragment)

typedef __attribute__((ext_vector_type(16))) __bf16 v16bf;
typedef __attribute__((ext_vector_type(8)))  float  v8f;

struct U32x8 { uint4 a, b; };   // 32 bytes -> one v16bf fragment

__device__ __forceinline__ unsigned short f2bf(float f) {
  unsigned u = __builtin_bit_cast(unsigned, f);
  u += 0x7FFFu + ((u >> 16) & 1u);            // round-to-nearest-even
  return (unsigned short)(u >> 16);
}

// ---------------------------------------------------------------------------
// Kernel 0a: Wb[h][k] = bf16(WT[k][h])   (transpose+convert, 32x32 LDS tiles)
// ---------------------------------------------------------------------------
__global__ __launch_bounds__(256)
void wt_to_bf16_T(const float* __restrict__ WT, unsigned short* __restrict__ Wb) {
  __shared__ float ts[32][33];
  const int h0 = blockIdx.x * 32;   // hidden tile
  const int k0 = blockIdx.y * 32;   // input-dim tile
  const int lr = threadIdx.x >> 5;  // 0..7
  const int lc = threadIdx.x & 31;
#pragma unroll
  for (int kk = 0; kk < 4; ++kk) {
    int k = k0 + lr + kk * 8;
    ts[lr + kk * 8][lc] = WT[(size_t)k * HID + h0 + lc];   // coalesced in h
  }
  __syncthreads();
#pragma unroll
  for (int kk = 0; kk < 4; ++kk) {
    int r = lr + kk * 8;   // local hidden row
    Wb[(size_t)(h0 + r) * IN + k0 + lc] = f2bf(ts[lc][r]); // coalesced in k
  }
}

// ---------------------------------------------------------------------------
// Kernel 0b: xcb = bf16(x - b2)
// ---------------------------------------------------------------------------
__global__ __launch_bounds__(256)
void make_xc_bf16(const float* __restrict__ x, const float* __restrict__ b2,
                  unsigned short* __restrict__ xcb) {
  int i = blockIdx.x * 256 + threadIdx.x;
  if (i < TOKENS * IN) {
    int c = i % IN;
    xcb[i] = f2bf(x[i] - b2[c]);
  }
}

// ---------------------------------------------------------------------------
// Kernel 1: fused encoder GEMM (bf16 WMMA) + exact per-token top-32
//   WG = 256 threads (8 waves), owns TT=64 tokens, loops over all 24576
//   hidden rows in chunks of 128 (16 rows per wave, 4 token-tiles per wave).
//   preacts = Wb @ xc^T.
// ---------------------------------------------------------------------------
#define XST 776            // padded bf16 stride of staged x tile
#define PST (TT + 1)       // padded f32 stride (tokens dim) of preact staging

__global__ __launch_bounds__(256)
void encode_topk(const unsigned short* __restrict__ Wb,
                 const unsigned short* __restrict__ xcb,
                 const float* __restrict__ b1,
                 float* __restrict__ topv, int* __restrict__ topi) {
  __shared__ __align__(16) unsigned short sx[TT * XST];  // 99,328 B
  __shared__ float sp[128 * PST];                        // 33,280 B

  const int tid  = threadIdx.x;
  const int lane = tid & 31;
  const int wv   = tid >> 5;           // wave 0..7
  const int tok0 = blockIdx.x * TT;

  // ---- stage TTx768 bf16 x-tile into LDS (vectorized, padded stride) ----
  for (int it = tid; it < TT * (IN / 8); it += 256) {
    int row = it / (IN / 8);
    int c8  = it % (IN / 8);
    uint4 v = *(const uint4*)(xcb + (size_t)(tok0 + row) * IN + c8 * 8);
    *(uint4*)(&sx[row * XST + c8 * 8]) = v;
  }
  __syncthreads();

  const int nloc   = lane & 15;   // A: row M / B: column N
  const int hihalf = lane >> 4;   // K-group select per WMMA bf16 layout

  // per-lane entry of the 8 owned tokens' top-32 lists (one entry per lane)
  float bestv[8]; int besti[8]; float curmin[8];
#pragma unroll
  for (int j = 0; j < 8; ++j) { bestv[j] = -3.0e38f; besti[j] = 0; curmin[j] = -3.0e38f; }

  for (int chunk = 0; chunk < HID / 128; ++chunk) {
    const int h0 = chunk * 128 + wv * 16;      // this wave's 16 hidden rows
    v8f acc[4];
#pragma unroll
    for (int q = 0; q < 4; ++q) acc[q] = (v8f){0.f,0.f,0.f,0.f,0.f,0.f,0.f,0.f};

    // A fragment base: lane holds row (h0+nloc), K groups {kh..kh+7, kh+16..kh+23}
    const unsigned short* arow = Wb + (size_t)(h0 + nloc) * IN + hihalf * 8;
    if (chunk + 1 < HID / 128)
      __builtin_prefetch(arow + 128 * IN, 0, 0);   // next chunk of Wb

#pragma unroll 2
    for (int kk = 0; kk < IN / 32; ++kk) {
      U32x8 ua;
      ua.a = *(const uint4*)(arow + kk * 32);
      ua.b = *(const uint4*)(arow + kk * 32 + 16);
      v16bf af = __builtin_bit_cast(v16bf, ua);
#pragma unroll
      for (int q = 0; q < 4; ++q) {
        const unsigned short* bp =
            &sx[(nloc + q * 16) * XST + kk * 32 + hihalf * 8];
        U32x8 ub;
        ub.a = *(const uint4*)bp;  ub.b = *(const uint4*)(bp + 16);
        v16bf bfq = __builtin_bit_cast(v16bf, ub);
        acc[q] = __builtin_amdgcn_wmma_f32_16x16x32_bf16(false, af, false, bfq,
                                                         (short)0, acc[q],
                                                         false, false);
      }
    }

    __syncthreads();   // previous iteration's selection finished reading sp
    // add b1 and stage 128xTT preact tile (C layout: M = v + 8*hihalf, N = nloc)
#pragma unroll
    for (int v = 0; v < 8; ++v) {
      int m = v + (hihalf << 3);
      float b = b1[h0 + m];
      float* row = &sp[(wv * 16 + m) * PST];
#pragma unroll
      for (int q = 0; q < 4; ++q) row[q * 16 + nloc] = acc[q][v] + b;
    }
    __syncthreads();

    // ---- running exact top-32: wave wv owns tokens wv*8 .. wv*8+7 ----
#pragma unroll
    for (int jj = 0; jj < 8; ++jj) {
      const int tloc = wv * 8 + jj;
      float sv[4]; unsigned mm[4];
#pragma unroll
      for (int i = 0; i < 4; ++i) {
        sv[i] = sp[(i * 32 + lane) * PST + tloc];
        mm[i] = (unsigned)__ballot(sv[i] > curmin[jj]);
      }
      const bool dirty = (mm[0] | mm[1] | mm[2] | mm[3]) != 0u;
#pragma unroll
      for (int i = 0; i < 4; ++i) {
        unsigned m = mm[i];
        while (m) {
          int b = __builtin_ctz(m); m &= m - 1;
          float cv = __shfl(sv[i], b, 32);
          int   ci = chunk * 128 + i * 32 + b;
          // tie-broken (value, lane) min reduction over the 32 list entries
          float mv = bestv[jj]; int ml = lane;
#pragma unroll
          for (int off = 16; off > 0; off >>= 1) {
            float ov = __shfl_xor(mv, off, 32);
            int   ol = __shfl_xor(ml, off, 32);
            if (ov < mv || (ov == mv && ol < ml)) { mv = ov; ml = ol; }
          }
          if (cv > mv && lane == ml) { bestv[jj] = cv; besti[jj] = ci; }
        }
      }
      if (dirty) {   // refresh cached threshold
        float mv = bestv[jj];
#pragma unroll
        for (int off = 16; off > 0; off >>= 1) {
          float ov = __shfl_xor(mv, off, 32);
          mv = ov < mv ? ov : mv;
        }
        curmin[jj] = mv;
      }
    }
  }

#pragma unroll
  for (int jj = 0; jj < 8; ++jj) {
    int t = tok0 + wv * 8 + jj;
    topv[(size_t)t * TOPK + lane] = bestv[jj];
    topi[(size_t)t * TOPK + lane] = besti[jj];
  }
}

// ---------------------------------------------------------------------------
// Kernel 2: decode  x_hat[t,:] = sum_k vals[t,k] * W[idx[t,k],:] + b2
// ---------------------------------------------------------------------------
__global__ __launch_bounds__(256)
void decode_k(const float* __restrict__ topv, const int* __restrict__ topi,
              const float* __restrict__ W, const float* __restrict__ b2,
              float* __restrict__ out) {
  __shared__ float svv[TOPK];
  __shared__ int   sii[TOPK];
  const int t   = blockIdx.x;
  const int tid = threadIdx.x;
  if (tid < TOPK) { svv[tid] = topv[(size_t)t * TOPK + tid];
                    sii[tid] = topi[(size_t)t * TOPK + tid]; }
  __syncthreads();
  float a0 = b2[tid], a1 = b2[tid + 256], a2 = b2[tid + 512];
#pragma unroll 8
  for (int k = 0; k < TOPK; ++k) {
    float v = svv[k];
    const float* wr = W + (size_t)sii[k] * IN;
    a0 += v * wr[tid]; a1 += v * wr[tid + 256]; a2 += v * wr[tid + 512];
  }
  float* o = out + (size_t)t * IN;
  o[tid] = a0; o[tid + 256] = a1; o[tid + 512] = a2;
}

// ---------------------------------------------------------------------------
extern "C" void kernel_launch(void* const* d_in, const int* in_sizes, int n_in,
                              void* d_out, int out_size, void* d_ws, size_t ws_size,
                              hipStream_t stream) {
  const float* x  = (const float*)d_in[0];
  const float* W  = (const float*)d_in[1];
  const float* WT = (const float*)d_in[2];
  const float* b1 = (const float*)d_in[3];
  const float* b2 = (const float*)d_in[4];
  float* out = (float*)d_out;

  // workspace layout (all offsets 256B-aligned)
  char* ws = (char*)d_ws;
  size_t offWb   = 0;
  size_t offXcb  = offWb  + (size_t)HID * IN * 2;      // 37,748,736
  size_t offTopv = offXcb + (size_t)TOKENS * IN * 2;   // +25,165,824
  size_t offTopi = offTopv + (size_t)TOKENS * TOPK * 4;
  unsigned short* Wb   = (unsigned short*)(ws + offWb);
  unsigned short* xcb  = (unsigned short*)(ws + offXcb);
  float*          topv = (float*)(ws + offTopv);
  int*            topi = (int*)(ws + offTopi);

  wt_to_bf16_T<<<dim3(HID / 32, IN / 32), 256, 0, stream>>>(WT, Wb);
  make_xc_bf16<<<(TOKENS * IN + 255) / 256, 256, 0, stream>>>(x, b2, xcb);
  encode_topk<<<TOKENS / TT, 256, 0, stream>>>(Wb, xcb, b1, topv, topi);
  decode_k<<<TOKENS, 256, 0, stream>>>(topv, topi, W, b2, out);

  (void)in_sizes; (void)n_in; (void)out_size; (void)ws_size;
}
